// Transformer_26517128085891
// MI455X (gfx1250) — matrix operand order repaired and checked by reading, not compiled, not run on hardware
//
#include <hip/hip_runtime.h>

#define B_  2
#define T_  1024
#define E_  512
#define H_  8
#define L_  4
#define NT_ 32000
#define FF_ 2048
#define HE_ (H_ * E_)
#define MROWS_ (B_ * T_)

typedef unsigned short u16;
typedef __attribute__((ext_vector_type(16))) __bf16 v16bf;
typedef __attribute__((ext_vector_type(8)))  float  v8f;

union FragBF { v16bf v; u16 u[16]; };

__device__ __forceinline__ u16 f2bf(float f) {
  unsigned int u = __float_as_uint(f);
  u += 0x7FFFu + ((u >> 16) & 1u);   // round-to-nearest-even
  return (u16)(u >> 16);
}

__device__ __forceinline__ float block_sum(float v, float* red) {
  const int tid = threadIdx.x;
  red[tid] = v;
  __syncthreads();
  for (int s = 128; s > 0; s >>= 1) {
    if (tid < s) red[tid] += red[tid + s];
    __syncthreads();
  }
  float r = red[0];
  __syncthreads();
  return r;
}

__device__ __forceinline__ float block_max(float v, float* red) {
  const int tid = threadIdx.x;
  red[tid] = v;
  __syncthreads();
  for (int s = 128; s > 0; s >>= 1) {
    if (tid < s) red[tid] = fmaxf(red[tid], red[tid + s]);
    __syncthreads();
  }
  float r = red[0];
  __syncthreads();
  return r;
}

// CDNA5 async memory->LDS copy (ASYNCcnt-tracked), 16B per lane.
__device__ __forceinline__ void async_copy_b128(unsigned lds_off, const u16* g) {
  asm volatile("global_load_async_to_lds_b128 %0, %1, off"
               :: "v"(lds_off), "v"((unsigned long long)(uintptr_t)g)
               : "memory");
}
__device__ __forceinline__ void wait_async0() {
  asm volatile("s_wait_asynccnt 0x0" ::: "memory");
}
__device__ __forceinline__ unsigned lds_addr32(const void* p) {
  // LDS flat addresses keep the LDS byte offset in addr[31:0] (ISA 10.2)
  return (unsigned)(uintptr_t)p;
}

// ---------------------------------------------------------------------------
// Generic batched strided bf16 WMMA GEMM: C = alpha * A @ B (+bias) (+relu)
//   A: (M x K) bf16, row stride lda
//   B: (K x N) bf16 row-major, or (N x K) row-major if FLAGS&GFLAG_BT
//   C: (M x N) f32, row stride ldc
// Block tile 128x128x32, 8 wave32 waves, each wave computes 32x64 via a
// 2x4 grid of v_wmma_f32_16x16x32_bf16 tiles. Double-buffered LDS with
// global_load_async_to_lds_b128 pipelining the next K-tile under the WMMAs.
// ---------------------------------------------------------------------------
#define BM 128
#define BN 128
#define BK 32
#define GFLAG_BT   1
#define GFLAG_RELU 2

template<int FLAGS>
__global__ __launch_bounds__(256)
void wmma_gemm_bf16(const u16* __restrict__ A, const u16* __restrict__ B,
                    float* __restrict__ C, const float* __restrict__ bias,
                    int Kdim,
                    long long lda, long long ldb, long long ldc,
                    long long sAb, long long sAh, long long sBb, long long sBh,
                    long long sCb, long long sCh, int nH,
                    float alpha)
{
  __shared__ u16 As[2][BM][BK + 8];   // row pitch 80B keeps b128 LDS alignment
  __shared__ u16 Bs[2][BN][BK + 8];   // stored K-transposed: Bs[buf][n][k]

  const int tid  = threadIdx.x;
  const int lane = tid & 31;
  const int wave = tid >> 5;
  const int r16  = lane & 15;
  const int g    = lane >> 4;
  const int wmRow = (wave & 3) * 32;
  const int wnCol = (wave >> 2) * 64;

  const int m0 = blockIdx.y * BM;
  const int n0 = blockIdx.x * BN;
  const int z  = blockIdx.z;
  const int bi = z / nH;
  const int hi = z - bi * nH;
  const long long aOff = (long long)bi * sAb + (long long)hi * sAh;
  const long long bOff = (long long)bi * sBb + (long long)hi * sBh;
  const long long cOff = (long long)bi * sCb + (long long)hi * sCh;

  // staging geometry: 2 threads/row, 16 halves (two b128) each
  const int rA = tid >> 1;
  const int cA = (tid & 1) * 16;
  const u16* aSrc = A + aOff + (long long)(m0 + rA) * lda + cA;   // +k0
  const u16* bSrcT = B + bOff + (long long)(n0 + rA) * ldb + cA;  // BT: +k0
  const int kkB = tid >> 3;            // 0..31 (non-BT transpose staging)
  const int cB  = (tid & 7) * 16;      // 0..112
  const u16* bSrcN = B + bOff + n0 + cB;                          // +(k0+kkB)*ldb

  unsigned aL0[2], aL1[2], bL0[2], bL1[2];
#pragma unroll
  for (int q = 0; q < 2; ++q) {
    aL0[q] = lds_addr32(&As[q][rA][cA]);
    aL1[q] = lds_addr32(&As[q][rA][cA + 8]);
    bL0[q] = lds_addr32(&Bs[q][rA][cA]);
    bL1[q] = lds_addr32(&Bs[q][rA][cA + 8]);
  }

  v8f acc[2][4];
#pragma unroll
  for (int mt = 0; mt < 2; ++mt)
#pragma unroll
    for (int nt = 0; nt < 4; ++nt)
      acc[mt][nt] = (v8f){0.f, 0.f, 0.f, 0.f, 0.f, 0.f, 0.f, 0.f};

  // ---- prologue: stage tile 0 into buffer 0 ----
  async_copy_b128(aL0[0], aSrc);
  async_copy_b128(aL1[0], aSrc + 8);
  if (FLAGS & GFLAG_BT) {
    async_copy_b128(bL0[0], bSrcT);
    async_copy_b128(bL1[0], bSrcT + 8);
  } else {
    const u16* bn = bSrcN + (long long)kkB * ldb;
    uint4 d0 = *(const uint4*)(bn);
    uint4 d1 = *(const uint4*)(bn + 8);
    u16 tmp[16];
    *(uint4*)(&tmp[0]) = d0;
    *(uint4*)(&tmp[8]) = d1;
#pragma unroll
    for (int i = 0; i < 16; ++i) Bs[0][cB + i][kkB] = tmp[i];
  }
  wait_async0();
  __syncthreads();

  for (int k0 = 0; k0 < Kdim; k0 += BK) {
    const int cur = (k0 >> 5) & 1;
    const int nxt = cur ^ 1;
    const bool more = (k0 + BK) < Kdim;

    // ---- kick off next tile (overlaps with WMMAs below) ----
    uint4 nd0 = {0, 0, 0, 0}, nd1 = {0, 0, 0, 0};
    if (more) {
      const u16* an = aSrc + (k0 + BK);
      async_copy_b128(aL0[nxt], an);
      async_copy_b128(aL1[nxt], an + 8);
      if (FLAGS & GFLAG_BT) {
        const u16* bn = bSrcT + (k0 + BK);
        async_copy_b128(bL0[nxt], bn);
        async_copy_b128(bL1[nxt], bn + 8);
      } else {
        const u16* bn = bSrcN + (long long)(k0 + BK + kkB) * ldb;
        nd0 = *(const uint4*)(bn);
        nd1 = *(const uint4*)(bn + 8);
        if (k0 + 2 * BK < Kdim)
          __builtin_prefetch((const void*)(bSrcN + (long long)(k0 + 2 * BK + kkB) * ldb), 0, 1);
      }
    }

    // ---- fragments from current buffer ----
    // ISA 16-bit A layout: per-lane contiguous K chunks of 8 at g*8 and 16+g*8;
// B fragment: same K chunks of 8 at g*8 and 16+g*8 for column r16.
    FragBF a[2], b[4];
#pragma unroll
    for (int mt = 0; mt < 2; ++mt) {
      const int row = wmRow + mt * 16 + r16;
      *(uint4*)(&a[mt].u[0]) = *(const uint4*)(&As[cur][row][g * 8]);
      *(uint4*)(&a[mt].u[8]) = *(const uint4*)(&As[cur][row][16 + g * 8]);
    }
#pragma unroll
    for (int nt = 0; nt < 4; ++nt) {
      const int col = wnCol + nt * 16 + r16;
      *(uint4*)(&b[nt].u[0]) = *(const uint4*)(&Bs[cur][col][g * 8]);
      *(uint4*)(&b[nt].u[8]) = *(const uint4*)(&Bs[cur][col][16 + g * 8]);
    }
#pragma unroll
    for (int mt = 0; mt < 2; ++mt)
#pragma unroll
      for (int nt = 0; nt < 4; ++nt)
        acc[mt][nt] = __builtin_amdgcn_wmma_f32_16x16x32_bf16(
            false, a[mt].v, false, b[nt].v, (short)0, acc[mt][nt], false, false);

    if (more) {
      if (!(FLAGS & GFLAG_BT)) {
        u16 tmp[16];
        *(uint4*)(&tmp[0]) = nd0;
        *(uint4*)(&tmp[8]) = nd1;
#pragma unroll
        for (int i = 0; i < 16; ++i) Bs[nxt][cB + i][kkB] = tmp[i];
      }
      wait_async0();
      __syncthreads();
    }
  }

  // ---- epilogue: C/D layout lane<16 -> M=i, lane>=16 -> M=8+i, N=lane&15 ----
#pragma unroll
  for (int mt = 0; mt < 2; ++mt) {
#pragma unroll
    for (int nt = 0; nt < 4; ++nt) {
      const int col = n0 + wnCol + nt * 16 + r16;
      const int rb  = m0 + wmRow + mt * 16 + g * 8;
      const float bv = bias ? bias[col] : 0.0f;
#pragma unroll
      for (int i = 0; i < 8; ++i) {
        float v = alpha * acc[mt][nt][i] + bv;
        if (FLAGS & GFLAG_RELU) v = fmaxf(v, 0.0f);
        C[cOff + (long long)(rb + i) * ldc + col] = v;
      }
    }
  }
}

// ---------------------------------------------------------------------------
__global__ __launch_bounds__(256)
void f32_to_bf16_kernel(const float* __restrict__ in, u16* __restrict__ out,
                        long long n)
{
  long long i = (long long)blockIdx.x * 256 + threadIdx.x;
  if (i < n) out[i] = f2bf(in[i]);
}

__global__ __launch_bounds__(256)
void embed_kernel(const int* __restrict__ x, const float* __restrict__ tok,
                  const float* __restrict__ pos, float* __restrict__ h,
                  u16* __restrict__ hbf)
{
  const int row = blockIdx.x;        // b*T + t
  const int t = row % T_;
  const int tk = x[row];
  for (int e = threadIdx.x; e < E_; e += 256) {
    float v = tok[(long long)tk * E_ + e] + pos[(long long)t * E_ + e];
    h  [(long long)row * E_ + e] = v;
    hbf[(long long)row * E_ + e] = f2bf(v);
  }
}

// LayerNorm(x + res) * g + b -> hout (f32) and hbf (bf16); one row per block.
__global__ __launch_bounds__(256)
void add_ln_kernel(const float* xin, const float* res,
                   const float* __restrict__ gg, const float* __restrict__ bb,
                   float* hout, u16* hbf)
{
  __shared__ float red[256];
  const long long row = blockIdx.x;
  const int tid = threadIdx.x;
  float v0 = xin[row * E_ + tid]       + res[row * E_ + tid];
  float v1 = xin[row * E_ + tid + 256] + res[row * E_ + tid + 256];
  float mean = block_sum(v0 + v1, red) * (1.0f / E_);
  float d0 = v0 - mean, d1 = v1 - mean;
  float var = block_sum(d0 * d0 + d1 * d1, red) * (1.0f / E_);
  float rstd = rsqrtf(var + 1e-5f);
  float o0 = d0 * rstd * gg[tid]       + bb[tid];
  float o1 = d1 * rstd * gg[tid + 256] + bb[tid + 256];
  hout[row * E_ + tid]       = o0;
  hout[row * E_ + tid + 256] = o1;
  hbf [row * E_ + tid]       = f2bf(o0);
  hbf [row * E_ + tid + 256] = f2bf(o1);
}

// Causal softmax over scores[z][row][:], writes bf16 probabilities.
__global__ __launch_bounds__(256)
void causal_softmax_kernel(const float* __restrict__ sc, u16* __restrict__ att)
{
  __shared__ float red[256];
  const int row = blockIdx.x;   // t
  const int z   = blockIdx.y;   // b*H + h
  const float* s = sc  + ((long long)z * T_ + row) * T_;
  u16*         a = att + ((long long)z * T_ + row) * T_;
  const int n = row + 1;
  float m = -3.402823466e38f;
  for (int j = threadIdx.x; j < n; j += 256) m = fmaxf(m, s[j]);
  m = block_max(m, red);
  float sum = 0.f;
  for (int j = threadIdx.x; j < n; j += 256) sum += __expf(s[j] - m);
  sum = block_sum(sum, red);
  const float inv = 1.0f / sum;
  for (int j = threadIdx.x; j < T_; j += 256)
    a[j] = (j < n) ? f2bf(__expf(s[j] - m) * inv) : (u16)0;
}

// emb_mean / emb_max over T, one thread per (b, e).
__global__ __launch_bounds__(256)
void meanmax_kernel(const float* __restrict__ h, float* __restrict__ out)
{
  const int idx = blockIdx.x * 256 + threadIdx.x;
  if (idx >= B_ * E_) return;
  const int b = idx / E_;
  const int e = idx - b * E_;
  float s = 0.f, mx = -3.402823466e38f;
  for (int t = 0; t < T_; ++t) {
    float v = h[((long long)b * T_ + t) * E_ + e];
    s += v;
    mx = fmaxf(mx, v);
  }
  const long long base = (long long)B_ * NT_ * T_;
  out[base + (long long)b * E_ + e] = s * (1.0f / T_);
  out[base + (long long)B_ * E_ + (long long)b * E_ + e] = mx;
}

// Faithful to torch: raw (B*T, NT) logits viewed as (B, NT, T); log_softmax
// over dim=1 normalizes stride-T slices of the flat buffer.
__global__ __launch_bounds__(256)
void log_softmax_kernel(const float* __restrict__ logits, float* __restrict__ out)
{
  __shared__ float red[256];
  const int t = blockIdx.x;
  const int b = blockIdx.y;
  const float* lf = logits + (long long)b * NT_ * T_ + t;
  float*       o  = out    + (long long)b * NT_ * T_ + t;
  float m = -3.402823466e38f;
  for (int n = threadIdx.x; n < NT_; n += 256)
    m = fmaxf(m, lf[(long long)n * T_]);
  m = block_max(m, red);
  float sum = 0.f;
  for (int n = threadIdx.x; n < NT_; n += 256)
    sum += __expf(lf[(long long)n * T_] - m);
  sum = block_sum(sum, red);
  const float lse = m + __logf(sum);
  for (int n = threadIdx.x; n < NT_; n += 256)
    o[(long long)n * T_] = lf[(long long)n * T_] - lse;
}

// ---------------------------------------------------------------------------
extern "C" void kernel_launch(void* const* d_in, const int* in_sizes, int n_in,
                              void* d_out, int out_size, void* d_ws, size_t ws_size,
                              hipStream_t stream)
{
  (void)in_sizes; (void)n_in; (void)out_size; (void)ws_size;

  const int*   x       = (const int*)  d_in[0];
  const float* tok_emb = (const float*)d_in[1];
  const float* pos_emb = (const float*)d_in[2];
  const float* Wq  = (const float*)d_in[3];
  const float* Wk  = (const float*)d_in[4];
  const float* Wv  = (const float*)d_in[5];
  const float* Wu  = (const float*)d_in[6];
  const float* bu  = (const float*)d_in[7];
  const float* g1  = (const float*)d_in[8];
  const float* b1n = (const float*)d_in[9];
  const float* g2  = (const float*)d_in[10];
  const float* b2n = (const float*)d_in[11];
  const float* W1  = (const float*)d_in[12];
  const float* bf1 = (const float*)d_in[13];
  const float* W2  = (const float*)d_in[14];
  const float* bf2 = (const float*)d_in[15];
  const float* Wp  = (const float*)d_in[16];
  const float* bp  = (const float*)d_in[17];
  float* out = (float*)d_out;

  // bump allocator over workspace
  char* wsb = (char*)d_ws;
  size_t off = 0;
  auto alloc = [&](size_t bytes) -> void* {
    off = (off + 255) & ~(size_t)255;
    void* p = wsb + off;
    off += bytes;
    return p;
  };

  u16* Wq_bf = (u16*)alloc((size_t)L_ * E_ * HE_ * 2);
  u16* Wk_bf = (u16*)alloc((size_t)L_ * E_ * HE_ * 2);
  u16* Wv_bf = (u16*)alloc((size_t)L_ * E_ * HE_ * 2);
  u16* Wu_bf = (u16*)alloc((size_t)L_ * HE_ * E_ * 2);
  u16* W1_bf = (u16*)alloc((size_t)L_ * E_ * FF_ * 2);
  u16* W2_bf = (u16*)alloc((size_t)L_ * FF_ * E_ * 2);
  u16* Wp_bf = (u16*)alloc((size_t)E_ * NT_ * 2);

  float* h      = (float*)alloc((size_t)MROWS_ * E_ * 4);
  u16*   h_bf   = (u16*)  alloc((size_t)MROWS_ * E_ * 2);
  u16*   q_bf   = (u16*)  alloc((size_t)MROWS_ * HE_ * 2);
  u16*   k_bf   = (u16*)  alloc((size_t)MROWS_ * HE_ * 2);
  u16*   v_bf   = (u16*)  alloc((size_t)MROWS_ * HE_ * 2);
  u16*   o_bf   = (u16*)  alloc((size_t)MROWS_ * HE_ * 2);
  u16*   f1_bf  = (u16*)  alloc((size_t)MROWS_ * FF_ * 2);
  u16*   att_bf = (u16*)  alloc((size_t)B_ * H_ * T_ * T_ * 2);
  float* scores = (float*)alloc((size_t)B_ * H_ * T_ * T_ * 4);
  float* tmpf   = (float*)alloc((size_t)MROWS_ * HE_ * 4);  // reused scratch
  float* logits = (float*)alloc((size_t)MROWS_ * NT_ * 4);

  auto conv = [&](const float* s, u16* d, long long n) {
    f32_to_bf16_kernel<<<dim3((unsigned)((n + 255) / 256)), dim3(256), 0, stream>>>(s, d, n);
  };
  auto gemm = [&](const u16* A, const u16* Bm, float* C, const float* bias,
                  int Md, int Nd, int Kd,
                  long long lda, long long ldb, long long ldc,
                  long long sAb, long long sAh, long long sBb, long long sBh,
                  long long sCb, long long sCh, int nH, int batches,
                  float alpha, int flags) {
    dim3 grid((unsigned)(Nd / BN), (unsigned)(Md / BM), (unsigned)batches);
    if (flags == GFLAG_BT)
      wmma_gemm_bf16<GFLAG_BT><<<grid, 256, 0, stream>>>(
          A, Bm, C, bias, Kd, lda, ldb, ldc, sAb, sAh, sBb, sBh, sCb, sCh, nH, alpha);
    else if (flags == GFLAG_RELU)
      wmma_gemm_bf16<GFLAG_RELU><<<grid, 256, 0, stream>>>(
          A, Bm, C, bias, Kd, lda, ldb, ldc, sAb, sAh, sBb, sBh, sCb, sCh, nH, alpha);
    else
      wmma_gemm_bf16<0><<<grid, 256, 0, stream>>>(
          A, Bm, C, bias, Kd, lda, ldb, ldc, sAb, sAh, sBb, sBh, sCb, sCh, nH, alpha);
  };

  // weights -> bf16 (every call; no cached state)
  conv(Wq, Wq_bf, (long long)L_ * E_ * HE_);
  conv(Wk, Wk_bf, (long long)L_ * E_ * HE_);
  conv(Wv, Wv_bf, (long long)L_ * E_ * HE_);
  conv(Wu, Wu_bf, (long long)L_ * HE_ * E_);
  conv(W1, W1_bf, (long long)L_ * E_ * FF_);
  conv(W2, W2_bf, (long long)L_ * FF_ * E_);
  conv(Wp, Wp_bf, (long long)E_ * NT_);

  embed_kernel<<<MROWS_, 256, 0, stream>>>(x, tok_emb, pos_emb, h, h_bf);

  const float inv_scale = 0.21022410f;  // 512^-0.25
  const long long sBH = (long long)H_ * T_ * T_;  // scores batch-b stride
  const long long sHH = (long long)T_ * T_;       // scores batch-h stride
  const long long qBB = (long long)T_ * HE_;      // q/k/v batch-b stride
  const long long qBH = (long long)E_;            // q/k/v batch-h stride

  for (int l = 0; l < L_; ++l) {
    const u16* wq = Wq_bf + (size_t)l * E_ * HE_;
    const u16* wk = Wk_bf + (size_t)l * E_ * HE_;
    const u16* wv = Wv_bf + (size_t)l * E_ * HE_;
    const u16* wu = Wu_bf + (size_t)l * HE_ * E_;
    const u16* w1 = W1_bf + (size_t)l * E_ * FF_;
    const u16* w2 = W2_bf + (size_t)l * FF_ * E_;

    // Q/K/V projections (scale folded into Q and K)
    gemm(h_bf, wq, tmpf, nullptr, MROWS_, HE_, E_, E_, HE_, HE_,
         0, 0, 0, 0, 0, 0, 1, 1, inv_scale, 0);
    conv(tmpf, q_bf, (long long)MROWS_ * HE_);
    gemm(h_bf, wk, tmpf, nullptr, MROWS_, HE_, E_, E_, HE_, HE_,
         0, 0, 0, 0, 0, 0, 1, 1, inv_scale, 0);
    conv(tmpf, k_bf, (long long)MROWS_ * HE_);
    gemm(h_bf, wv, tmpf, nullptr, MROWS_, HE_, E_, E_, HE_, HE_,
         0, 0, 0, 0, 0, 0, 1, 1, 1.0f, 0);
    conv(tmpf, v_bf, (long long)MROWS_ * HE_);

    // scores[b,h] = Q_bh @ K_bh^T  (batched over b*H, B given N x K)
    gemm(q_bf, k_bf, scores, nullptr, T_, T_, E_, HE_, HE_, T_,
         qBB, qBH, qBB, qBH, sBH, sHH, H_, B_ * H_, 1.0f, GFLAG_BT);

    causal_softmax_kernel<<<dim3(T_, B_ * H_), 256, 0, stream>>>(scores, att_bf);

    // O[b,h] = att @ V_bh  -> (B,T,H,E) layout
    gemm(att_bf, v_bf, tmpf, nullptr, T_, E_, T_, T_, HE_, HE_,
         sBH, sHH, qBB, qBH, qBB, qBH, H_, B_ * H_, 1.0f, 0);
    conv(tmpf, o_bf, (long long)MROWS_ * HE_);

    // unify heads + bias, then residual + LN1
    gemm(o_bf, wu, tmpf, bu + (size_t)l * E_, MROWS_, E_, HE_, HE_, E_, E_,
         0, 0, 0, 0, 0, 0, 1, 1, 1.0f, 0);
    add_ln_kernel<<<MROWS_, 256, 0, stream>>>(tmpf, h, g1 + (size_t)l * E_,
                                              b1n + (size_t)l * E_, h, h_bf);

    // FF: relu(h @ W1 + b) @ W2 + b, residual + LN2
    gemm(h_bf, w1, tmpf, bf1 + (size_t)l * FF_, MROWS_, FF_, E_, E_, FF_, FF_,
         0, 0, 0, 0, 0, 0, 1, 1, 1.0f, GFLAG_RELU);
    conv(tmpf, f1_bf, (long long)MROWS_ * FF_);
    gemm(f1_bf, w2, tmpf, bf2 + (size_t)l * E_, MROWS_, E_, FF_, FF_, E_, E_,
         0, 0, 0, 0, 0, 0, 1, 1, 1.0f, 0);
    add_ln_kernel<<<MROWS_, 256, 0, stream>>>(tmpf, h, g2 + (size_t)l * E_,
                                              b2n + (size_t)l * E_, h, h_bf);
  }

  meanmax_kernel<<<dim3((B_ * E_ + 255) / 256), 256, 0, stream>>>(h, out);

  // vocab projection + bias -> raw logits (B*T, NT)
  gemm(h_bf, Wp_bf, logits, bp, MROWS_, NT_, E_, E_, NT_, NT_,
       0, 0, 0, 0, 0, 0, 1, 1, 1.0f, 0);

  log_softmax_kernel<<<dim3(T_, B_), 256, 0, stream>>>(logits, out);
}
